// XFormerAttention_83451214561794
// MI455X (gfx1250) — matrix operand order repaired and checked by reading, not compiled, not run on hardware
//
#include <hip/hip_runtime.h>
#include <hip/hip_bf16.h>

// ---------------------------------------------------------------------------
// MHA forward for MI455X (gfx1250, wave32, WMMA f32_16x16x32_f16)
//   B=2, T=2048, D=512, H=8, hd=64
// Pipeline:
//   1) qk_proj (x2): (X @ W^T + b) * alpha, fp32->f16 -> [B,H,T,hd]
//      v_proj  (x1): (X @ W^T + b),         fp32->f16 -> [B,H,hd,T] (transposed,
//                    packed b128 epilogue stores; PV B-frags become contiguous)
//   2) flash_attn: online-softmax, 16 q-rows per wave, WMMA for S and PV
//   3) out_proj: attn(f16) @ Wo^T + bo -> fp32
// ---------------------------------------------------------------------------

#define B_SZ   2
#define T_SEQ  2048
#define D_MOD  512
#define NHEAD  8
#define HDIM   64
#define NEGBIG (-10000.0f)

typedef _Float16 v4h  __attribute__((ext_vector_type(4)));
typedef _Float16 v8h  __attribute__((ext_vector_type(8)));
typedef _Float16 v16h __attribute__((ext_vector_type(16)));
typedef float    v4f  __attribute__((ext_vector_type(4)));
typedef float    v8f  __attribute__((ext_vector_type(8)));

__device__ __forceinline__ v8f wmma_f16(v16h a, v16h b, v8f c) {
  // D = A(16x32 f16) x B(32x16 f16) + C(16x16 f32)
  return __builtin_amdgcn_wmma_f32_16x16x32_f16(
      /*neg_a=*/false, a, /*neg_b=*/false, b,
      /*c_mod=*/(short)0, c, /*reuse_a=*/false, /*reuse_b=*/false);
}

__device__ __forceinline__ v16h cat8(v8h lo, v8h hi) {
  return __builtin_shufflevector(lo, hi, 0,1,2,3,4,5,6,7,8,9,10,11,12,13,14,15);
}

// A fragment: 16x32 (MxK) f16, row-major source with stride ld (halves).
// ISA layout: lane<16 -> K {0..7,16..23}; lane>=16 -> K {8..15,24..31}; M=lane&15
__device__ __forceinline__ v16h load_frag_a(const _Float16* p, int ld, int lane) {
  const int row = lane & 15;
  const int kb  = (lane >> 4) << 3;               // 0 or 8
  const _Float16* r0 = p + row * ld + kb;
  return cat8(*(const v8h*)(r0), *(const v8h*)(r0 + 16));
}

// B fragment where source tile is stored [N][K] row-major (i.e. B = tile^T):
// lane: n = lane&15; lane<16 -> K {0..7,16..23}; lane>=16 -> K {8..15,24..31}
__device__ __forceinline__ v16h load_frag_b_nk(const _Float16* p, int ld, int lane) {
  const int n  = lane & 15;
  const int kb = (lane >> 4) << 3;                // 0 or 8
  const _Float16* r0 = p + n * ld + kb;
  return cat8(*(const v8h*)(r0), *(const v8h*)(r0 + 16));
}

__device__ __forceinline__ v4h cvt4(v4f f) {
  v4h h;
#pragma unroll
  for (int j = 0; j < 4; ++j) h[j] = (_Float16)f[j];
  return h;
}

// Shared 64x64x512 mixed-precision GEMM body: acc[2][2] per wave.
// sA/sB: [64][40] f16 staging tiles. X, W are fp32 [.,512] row-major.
__device__ __forceinline__ void gemm_body_f32(const float* __restrict__ X,
                                              const float* __restrict__ W,
                                              int m0, int n0, int tid, int lane,
                                              int wm, int wn,
                                              _Float16 (*sA)[40], _Float16 (*sB)[40],
                                              v8f acc[2][2])
{
  for (int kt = 0; kt < D_MOD; kt += 32) {
#pragma unroll
    for (int i = 0; i < 4; ++i) {
      const int fi = tid + i * 128;               // float4 index in [0, 512)
      const int r  = fi >> 3;                     // 8 float4 per 32-wide row
      const int c4 = (fi & 7) * 4;
      const v4f xa = *(const v4f*)(X + (size_t)(m0 + r) * D_MOD + kt + c4);
      const v4f xb = *(const v4f*)(W + (size_t)(n0 + r) * D_MOD + kt + c4);
      *(v4h*)(&sA[r][c4]) = cvt4(xa);
      *(v4h*)(&sB[r][c4]) = cvt4(xb);
    }
    __syncthreads();
    const v16h a0 = load_frag_a(&sA[wm][0],       40, lane);
    const v16h a1 = load_frag_a(&sA[wm + 16][0],  40, lane);
    const v16h b0 = load_frag_b_nk(&sB[wn][0],      40, lane);
    const v16h b1 = load_frag_b_nk(&sB[wn + 16][0], 40, lane);
    acc[0][0] = wmma_f16(a0, b0, acc[0][0]);
    acc[0][1] = wmma_f16(a0, b1, acc[0][1]);
    acc[1][0] = wmma_f16(a1, b0, acc[1][0]);
    acc[1][1] = wmma_f16(a1, b1, acc[1][1]);
    __syncthreads();
  }
}

// ---------------------------------------------------------------------------
// Q/K projection -> [B,H,T,hd] f16 (alpha folds 1/sqrt(hd) for Q)
// ---------------------------------------------------------------------------
__global__ __launch_bounds__(128)
void qk_proj_kernel(const float* __restrict__ X, const float* __restrict__ W,
                    const float* __restrict__ bias, _Float16* __restrict__ out,
                    float alpha)
{
  __shared__ __align__(16) _Float16 sA[64][40];
  __shared__ __align__(16) _Float16 sB[64][40];
  const int tid  = threadIdx.x;
  const int lane = tid & 31;
  const int wave = tid >> 5;
  const int m0 = blockIdx.x * 64, n0 = blockIdx.y * 64;
  const int wm = (wave >> 1) * 32, wn = (wave & 1) * 32;

  const v8f z8 = {0.f,0.f,0.f,0.f,0.f,0.f,0.f,0.f};
  v8f acc[2][2] = {{z8, z8}, {z8, z8}};
  gemm_body_f32(X, W, m0, n0, tid, lane, wm, wn, sA, sB, acc);

  const int rgrp = lane >> 4;
  const int nlane = lane & 15;
#pragma unroll
  for (int mi = 0; mi < 2; ++mi)
#pragma unroll
    for (int ni = 0; ni < 2; ++ni) {
      const int col = n0 + wn + ni * 16 + nlane;          // model dim
      const int h   = col >> 6;
      const int dd  = col & (HDIM - 1);
      const float bs = bias[col];
#pragma unroll
      for (int r = 0; r < 8; ++r) {
        const int row = m0 + wm + mi * 16 + rgrp * 8 + r; // token row in [0, B*T)
        const int bb = row >> 11;                         // / T_SEQ
        const int t  = row & (T_SEQ - 1);
        const size_t headbase = (size_t)(bb * NHEAD + h) * T_SEQ * HDIM;
        out[headbase + (size_t)t * HDIM + dd] = (_Float16)((acc[mi][ni][r] + bs) * alpha);
      }
    }
}

// ---------------------------------------------------------------------------
// V projection -> [B,H,hd,T] f16 (transposed per head, packed b128 stores:
// a lane's 8 accumulator rows are contiguous along T for fixed dd)
// ---------------------------------------------------------------------------
__global__ __launch_bounds__(128)
void v_proj_kernel(const float* __restrict__ X, const float* __restrict__ W,
                   const float* __restrict__ bias, _Float16* __restrict__ out)
{
  __shared__ __align__(16) _Float16 sA[64][40];
  __shared__ __align__(16) _Float16 sB[64][40];
  const int tid  = threadIdx.x;
  const int lane = tid & 31;
  const int wave = tid >> 5;
  const int m0 = blockIdx.x * 64, n0 = blockIdx.y * 64;
  const int wm = (wave >> 1) * 32, wn = (wave & 1) * 32;

  const v8f z8 = {0.f,0.f,0.f,0.f,0.f,0.f,0.f,0.f};
  v8f acc[2][2] = {{z8, z8}, {z8, z8}};
  gemm_body_f32(X, W, m0, n0, tid, lane, wm, wn, sA, sB, acc);

  const int rgrp = lane >> 4;
  const int nlane = lane & 15;
#pragma unroll
  for (int mi = 0; mi < 2; ++mi)
#pragma unroll
    for (int ni = 0; ni < 2; ++ni) {
      const int col = n0 + wn + ni * 16 + nlane;
      const int h   = col >> 6;
      const int dd  = col & (HDIM - 1);
      const float bs = bias[col];
      const int row0 = m0 + wm + mi * 16 + rgrp * 8;      // 8 consecutive tokens
      const int bb = row0 >> 11;
      const int t0 = row0 & (T_SEQ - 1);
      v8h pack;
#pragma unroll
      for (int r = 0; r < 8; ++r) pack[r] = (_Float16)(acc[mi][ni][r] + bs);
      const size_t headbase = (size_t)(bb * NHEAD + h) * T_SEQ * HDIM;
      *(v8h*)(out + headbase + (size_t)dd * T_SEQ + t0) = pack;  // 16B store
    }
}

// ---------------------------------------------------------------------------
// Flash attention: grid (T/64, B*H); 4 waves/block, 16 q-rows per wave.
// ---------------------------------------------------------------------------
__global__ __launch_bounds__(128)
void flash_attn_kernel(const _Float16* __restrict__ Qh,   // [B,H,T,hd]
                       const _Float16* __restrict__ Kh,   // [B,H,T,hd]
                       const _Float16* __restrict__ Vt,   // [B,H,hd,T]
                       const unsigned char* __restrict__ kpm,
                       _Float16* __restrict__ attn)       // [B,T,D]
{
  __shared__ __align__(16) _Float16 pbuf[4][16][32];  // per-wave P tile
  const int tid  = threadIdx.x;
  const int lane = tid & 31;
  const int wave = tid >> 5;
  const int bh = blockIdx.y;
  const int b  = bh >> 3;          // / NHEAD
  const int h  = bh & (NHEAD - 1);
  const int qbase = blockIdx.x * 64 + wave * 16;
  const _Float16* Qp = Qh + (size_t)bh * T_SEQ * HDIM;
  const _Float16* Kp = Kh + (size_t)bh * T_SEQ * HDIM;
  const _Float16* Vp = Vt + (size_t)bh * T_SEQ * HDIM;   // [hd][T]
  const int rgrp  = lane >> 4;     // 0: rows 0-7, 1: rows 8-15 (C/D layout)
  const int nlane = lane & 15;

  // Q fragments for this wave's 16 rows; hd=64 -> two K=32 chunks
  v16h aq[2];
  {
    const _Float16* q0 = Qp + (size_t)(qbase + nlane) * HDIM + rgrp * 8;
#pragma unroll
    for (int f = 0; f < 2; ++f)
      aq[f] = cat8(*(const v8h*)(q0 + f * 32), *(const v8h*)(q0 + f * 32 + 16));
  }

  const v8f z8 = {0.f,0.f,0.f,0.f,0.f,0.f,0.f,0.f};
  v8f acc[4] = {z8, z8, z8, z8};
  float mrun[8], lrun[8];
#pragma unroll
  for (int r = 0; r < 8; ++r) { mrun[r] = -3.0e38f; lrun[r] = 0.0f; }

  const int kend = qbase + 16;     // causal: keys <= qbase+15
  for (int kb = 0; kb < kend; kb += 32) {
    // prefetch next K / V^T blocks while this one computes
    if (kb + 32 < kend) {
      __builtin_prefetch(Kp + (size_t)(kb + 32 + lane) * HDIM, 0, 3);
      __builtin_prefetch(Vp + (size_t)lane * T_SEQ + kb + 32, 0, 3);
    }
    float s[2][8];
#pragma unroll
    for (int j = 0; j < 2; ++j) {
      const int key0 = kb + j * 16;
      // K^T fragments (B(k,n) = K[key0+n][k]) - same K order as A fragments
      const _Float16* kr = Kp + (size_t)(key0 + nlane) * HDIM + rgrp * 8;
      const v16h bk0 = cat8(*(const v8h*)(kr),      *(const v8h*)(kr + 16));
      const v16h bk1 = cat8(*(const v8h*)(kr + 32), *(const v8h*)(kr + 48));
      v8f c = z8;
      c = wmma_f16(aq[0], bk0, c);
      c = wmma_f16(aq[1], bk1, c);
      const int col = key0 + nlane;
      const int padded = (int)kpm[(size_t)b * T_SEQ + col];
#pragma unroll
      for (int r = 0; r < 8; ++r) {
        const int row = qbase + rgrp * 8 + r;
        float v = fminf(fmaxf(c[r], NEGBIG), -NEGBIG);  // reference clamp
        if ((col > row) | padded) v = NEGBIG;           // masked -> -1e4 exactly
        s[j][r] = v;
      }
    }
    // online softmax over these 32 columns (each row lives in one 16-lane half)
#pragma unroll
    for (int r = 0; r < 8; ++r) {
      float lm = fmaxf(s[0][r], s[1][r]);
#pragma unroll
      for (int off = 8; off >= 1; off >>= 1)
        lm = fmaxf(lm, __shfl_xor(lm, off, 16));
      const float mnew = fmaxf(mrun[r], lm);
      const float corr = __expf(mrun[r] - mnew);
      const float p0 = __expf(s[0][r] - mnew);
      const float p1 = __expf(s[1][r] - mnew);
      float ls = p0 + p1;
#pragma unroll
      for (int off = 8; off >= 1; off >>= 1)
        ls += __shfl_xor(ls, off, 16);
      lrun[r] = lrun[r] * corr + ls;
      mrun[r] = mnew;
#pragma unroll
      for (int n = 0; n < 4; ++n) acc[n][r] *= corr;
      pbuf[wave][rgrp * 8 + r][nlane]      = (_Float16)p0;
      pbuf[wave][rgrp * 8 + r][16 + nlane] = (_Float16)p1;
    }
    // LDS ops from one wave complete in order (DScnt in-order); just stop the
    // compiler from reordering across the store->load sequence.
    asm volatile("" ::: "memory");
    const v16h ap = load_frag_a(&pbuf[wave][0][0], 32, lane);
    // V^T fragments: [N][K] layout -> contiguous 16-byte loads per lane
#pragma unroll
    for (int n = 0; n < 4; ++n) {
      const v16h bv = load_frag_b_nk(Vp + (size_t)(n * 16) * T_SEQ + kb, T_SEQ, lane);
      acc[n] = wmma_f16(ap, bv, acc[n]);
    }
  }

  // normalize, write [B, T, D] f16 (heads re-interleaved)
#pragma unroll
  for (int r = 0; r < 8; ++r) {
    const float inv = 1.0f / lrun[r];
    const int row = qbase + rgrp * 8 + r;
    const size_t base = ((size_t)b * T_SEQ + row) * D_MOD + h * HDIM;
#pragma unroll
    for (int n = 0; n < 4; ++n)
      attn[base + n * 16 + nlane] = (_Float16)(acc[n][r] * inv);
  }
}

// ---------------------------------------------------------------------------
// Output projection: out = attn(f16) @ Wo^T + bo, fp32 result.
// ---------------------------------------------------------------------------
__global__ __launch_bounds__(128)
void out_proj_kernel(const _Float16* __restrict__ A, const float* __restrict__ W,
                     const float* __restrict__ bias, float* __restrict__ out)
{
  __shared__ __align__(16) _Float16 sA[64][40];
  __shared__ __align__(16) _Float16 sB[64][40];
  const int tid  = threadIdx.x;
  const int lane = tid & 31;
  const int wave = tid >> 5;
  const int m0 = blockIdx.x * 64, n0 = blockIdx.y * 64;
  const int wm = (wave >> 1) * 32, wn = (wave & 1) * 32;

  const v8f z8 = {0.f,0.f,0.f,0.f,0.f,0.f,0.f,0.f};
  v8f acc[2][2] = {{z8, z8}, {z8, z8}};

  for (int kt = 0; kt < D_MOD; kt += 32) {
    // A tile is already f16: two b128 loads per thread
#pragma unroll
    for (int i = 0; i < 2; ++i) {
      const int ui = tid + i * 128;               // v8h unit in [0, 256)
      const int r  = ui >> 2;                     // 4 units per 32-wide row
      const int c8 = (ui & 3) * 8;
      *(v8h*)(&sA[r][c8]) = *(const v8h*)(A + (size_t)(m0 + r) * D_MOD + kt + c8);
    }
    // W tile fp32 -> f16
#pragma unroll
    for (int i = 0; i < 4; ++i) {
      const int fi = tid + i * 128;
      const int r  = fi >> 3;
      const int c4 = (fi & 7) * 4;
      const v4f xb = *(const v4f*)(W + (size_t)(n0 + r) * D_MOD + kt + c4);
      *(v4h*)(&sB[r][c4]) = cvt4(xb);
    }
    __syncthreads();
    const v16h a0 = load_frag_a(&sA[wm][0],       40, lane);
    const v16h a1 = load_frag_a(&sA[wm + 16][0],  40, lane);
    const v16h b0 = load_frag_b_nk(&sB[wn][0],      40, lane);
    const v16h b1 = load_frag_b_nk(&sB[wn + 16][0], 40, lane);
    acc[0][0] = wmma_f16(a0, b0, acc[0][0]);
    acc[0][1] = wmma_f16(a0, b1, acc[0][1]);
    acc[1][0] = wmma_f16(a1, b0, acc[1][0]);
    acc[1][1] = wmma_f16(a1, b1, acc[1][1]);
    __syncthreads();
  }

  const int rgrp = lane >> 4;
  const int nlane = lane & 15;
#pragma unroll
  for (int mi = 0; mi < 2; ++mi)
#pragma unroll
    for (int ni = 0; ni < 2; ++ni) {
      const int col = n0 + wn + ni * 16 + nlane;
      const float bs = bias[col];
#pragma unroll
      for (int r = 0; r < 8; ++r) {
        const int row = m0 + wm + mi * 16 + rgrp * 8 + r;
        out[(size_t)row * D_MOD + col] = acc[mi][ni][r] + bs;
      }
    }
}

// ---------------------------------------------------------------------------
extern "C" void kernel_launch(void* const* d_in, const int* in_sizes, int n_in,
                              void* d_out, int out_size, void* d_ws, size_t ws_size,
                              hipStream_t stream) {
  (void)in_sizes; (void)n_in; (void)out_size; (void)ws_size;

  const float* query = (const float*)d_in[0];
  const float* key   = (const float*)d_in[1];
  const float* value = (const float*)d_in[2];
  const unsigned char* kpm = (const unsigned char*)d_in[3];  // bool [B,T]
  const float* Wq = (const float*)d_in[4];  const float* bq = (const float*)d_in[5];
  const float* Wk = (const float*)d_in[6];  const float* bk = (const float*)d_in[7];
  const float* Wv = (const float*)d_in[8];  const float* bv = (const float*)d_in[9];
  const float* Wo = (const float*)d_in[10]; const float* bo = (const float*)d_in[11];
  float* out = (float*)d_out;

  const size_t per = (size_t)B_SZ * T_SEQ * D_MOD;  // elements per f16 tensor
  _Float16* Qh = (_Float16*)d_ws;                   // [B,H,T,hd] f16
  _Float16* Kh = Qh + per;
  _Float16* Vh = Kh + per;                          // [B,H,hd,T] f16 (transposed)
  _Float16* At = Vh + per;                          // [B,T,D]   f16

  const dim3 gProj(B_SZ * T_SEQ / 64, D_MOD / 64);
  // fold 1/sqrt(hd) into Q
  qk_proj_kernel<<<gProj, 128, 0, stream>>>(query, Wq, bq, Qh, 0.125f);
  qk_proj_kernel<<<gProj, 128, 0, stream>>>(key,   Wk, bk, Kh, 1.0f);
  v_proj_kernel <<<gProj, 128, 0, stream>>>(value, Wv, bv, Vh);

  flash_attn_kernel<<<dim3(T_SEQ / 64, B_SZ * NHEAD), 128, 0, stream>>>(Qh, Kh, Vh, kpm, At);

  out_proj_kernel<<<dim3(B_SZ * T_SEQ / 64, D_MOD / 64), 128, 0, stream>>>(At, Wo, bo, out);
}
